// MentionDetectorCore_163208757712
// MI455X (gfx1250) — compile-verified, matrix operand order checked by reading, not run on hardware
//
#include <hip/hip_runtime.h>
#include <hip/hip_bf16.h>

typedef __attribute__((ext_vector_type(16))) _Float16 v16h;
typedef __attribute__((ext_vector_type(8)))  _Float16 v8h;
typedef __attribute__((ext_vector_type(4)))  _Float16 h4;
typedef __attribute__((ext_vector_type(8)))  float    v8f;
typedef __attribute__((ext_vector_type(4)))  unsigned int v4u;
typedef __attribute__((ext_vector_type(8)))  int      v8i_t;
typedef __attribute__((ext_vector_type(4)))  int      v4i_t;

#define HDIM 1024
#define HD   512
#define SEQ  512
#define BATCH 2
#define MTOT (BATCH*SEQ)   // 1024 rows of flattened emb

// ---------------------------------------------------------------------------
// TDM helper: issue a 2D tensor_load_to_lds.
//   Tile rows land contiguously in LDS with pad_amount DWORDs inserted every
//   pad_interval DWORDs: 16 DW data + 4 DW pad  ->  row pitch 40 halves.
// D# packing per CDNA5 ISA 8.3/8.4 (group0: count/lds/global/type,
// group1: data_size, pad, dims, strides). 6-arg clang-23 builtin form.
// ---------------------------------------------------------------------------
__device__ __forceinline__ void tdm_load_2d(unsigned int lds_off,
                                            const void* gaddr,
                                            unsigned int td0, unsigned int td1,
                                            unsigned int tile0, unsigned int tile1,
                                            unsigned int stride0)
{
    unsigned long long a = (unsigned long long)gaddr;
    v4u g0;
    g0[0] = 1u;                                        // count=1, user mode
    g0[1] = lds_off;                                   // lds_addr
    g0[2] = (unsigned int)a;                           // global_addr[31:0]
    g0[3] = (unsigned int)(a >> 32) | (2u << 30);      // addr[56:32] | type=2

    // data_size=1 (2B), pad_enable=1, pad_interval=3 (16 DW), pad_amount=3 (4 DW)
    const unsigned int w0 = (1u << 16) | (1u << 20) | (3u << 22) | (3u << 25);
    v8i_t g1;
    g1[0] = (int)w0;
    g1[1] = (int)((td0 & 0xFFFFu) << 16);                       // tensor_dim0 lo
    g1[2] = (int)((td0 >> 16) | ((td1 & 0xFFFFu) << 16));       // td0 hi | td1 lo
    g1[3] = (int)((td1 >> 16) | (tile0 << 16));                 // td1 hi | tile_dim0
    g1[4] = (int)(tile1 & 0xFFFFu);                             // tile_dim1 (dim2=0)
    g1[5] = (int)stride0;                                       // dim0_stride lo32
    g1[6] = 0;                                                  // stride hi | dim1_stride lo
    g1[7] = 0;

    v4i_t z4 = {0, 0, 0, 0};
    v8i_t z8 = {0, 0, 0, 0, 0, 0, 0, 0};
    __builtin_amdgcn_tensor_load_to_lds(g0, g1, z4, z4, z8, 0);
}

// ---------------------------------------------------------------------------
// Kernel 0a: straight f32 -> f16 conversion (emb), 4 elems/thread
// ---------------------------------------------------------------------------
__global__ __launch_bounds__(256)
void cvt_f32_to_f16(const float* __restrict__ src, _Float16* __restrict__ dst, int n4) {
    int i = blockIdx.x * blockDim.x + threadIdx.x;
    if (i < n4) {
        float4 f = ((const float4*)src)[i];
        h4 o;
        o[0] = (_Float16)f.x; o[1] = (_Float16)f.y;
        o[2] = (_Float16)f.z; o[3] = (_Float16)f.w;
        *(h4*)&dst[i * 4] = o;
    }
}

// ---------------------------------------------------------------------------
// Kernel 0b: f32 (K x N) -> f16 transposed (N x K), 32x32 LDS tile.
// Pre-transposing the weights lets the TDM stage B tiles in [n][k] order
// with no per-element LDS scatter in the GEMM hot loop.
// ---------------------------------------------------------------------------
__global__ __launch_bounds__(256)
void cvt_transpose_f16(const float* __restrict__ src, _Float16* __restrict__ dst,
                       int K, int N)
{
    __shared__ float st[32][33];
    const int tid = threadIdx.x;
    const int kb = blockIdx.x * 32, nb = blockIdx.y * 32;
    const int r = tid >> 3;          // 0..31
    const int c4 = (tid & 7) * 4;    // 0..28

    float4 v = *(const float4*)&src[(size_t)(kb + r) * N + nb + c4];
    st[r][c4 + 0] = v.x; st[r][c4 + 1] = v.y;
    st[r][c4 + 2] = v.z; st[r][c4 + 3] = v.w;
    __syncthreads();

    h4 o;
    #pragma unroll
    for (int q = 0; q < 4; ++q) o[q] = (_Float16)st[c4 + q][r];
    *(h4*)&dst[(size_t)(nb + r) * K + kb + c4] = o;
}

// ---------------------------------------------------------------------------
// Kernel 1: fused triple-projection WMMA GEMM, TDM-staged, double buffered.
//   mat 0: h_s = relu(emb @ W_s1 + b_s1)
//   mat 1: A   = emb @ W_e1[:H] + b_e1   (bias folded)
//   mat 2: C   = emb @ W_e1[H:]
// Block tile 128x64, 8 waves (4x2), wave tile 32x32 -> 4 WMMAs/K-step.
// ---------------------------------------------------------------------------
__global__ __launch_bounds__(256)
void fused_proj_wmma(const _Float16* __restrict__ embh,      // [1024][1024] f16
                     const _Float16* __restrict__ wsT,       // W_s1^T  [512][1024] f16
                     const _Float16* __restrict__ weTs,      // W1_start^T [512][1024]
                     const _Float16* __restrict__ weTe,      // W1_end^T   [512][1024]
                     const float* __restrict__ b_s1,
                     const float* __restrict__ b_e1,
                     float* __restrict__ hs,
                     float* __restrict__ Ap,
                     float* __restrict__ Cm)
{
    __shared__ _Float16 As[2][128][40];   // TDM-padded pitch: 32 data + 8 pad halves
    __shared__ _Float16 Bs[2][64][40];

    const int tid  = threadIdx.x;
    const int lane = tid & 31;
    const int w    = tid >> 5;       // wave 0..7
    const int wm   = w & 3;          // 0..3 : 32-row slice
    const int wn   = w >> 2;         // 0..1 : 32-col slice
    const int half = lane >> 4;
    const int lrow = lane & 15;

    const int mBase = blockIdx.x * 128;
    const int gy    = blockIdx.y;            // 0..23
    const int mat   = gy >> 3;               // 0..2
    const int nBase = (gy & 7) * 64;

    const _Float16* Wsel = (mat == 0) ? wsT : (mat == 1 ? weTs : weTe);
    const float*    bias = (mat == 0) ? b_s1 : (mat == 1 ? b_e1 : nullptr);
    float*          outp = (mat == 0) ? hs : (mat == 1 ? Ap : Cm);

    const unsigned int ldsA0 = (unsigned int)(size_t)&As[0][0][0];
    const unsigned int ldsA1 = (unsigned int)(size_t)&As[1][0][0];
    const unsigned int ldsB0 = (unsigned int)(size_t)&Bs[0][0][0];
    const unsigned int ldsB1 = (unsigned int)(size_t)&Bs[1][0][0];

    v8f acc00 = {}, acc01 = {}, acc10 = {}, acc11 = {};

    const int NSTEP = HDIM / 32;     // 32 K-steps

    // prologue: DMA tile pair for k0=0 into buffer 0 (wave 0 only)
    if (w == 0) {
        tdm_load_2d(ldsA0, embh + (size_t)mBase * HDIM, HDIM, HDIM, 32, 128, HDIM);
        tdm_load_2d(ldsB0, Wsel + (size_t)nBase * HDIM, HDIM, HD,   32, 64,  HDIM);
    }

    for (int it = 0; it < NSTEP; ++it) {
        const int buf = it & 1;
        if (w == 0) {
            if (it + 1 < NSTEP) {
                const int k1 = (it + 1) * 32;
                tdm_load_2d(buf ? ldsA0 : ldsA1,
                            embh + (size_t)mBase * HDIM + k1, HDIM, HDIM, 32, 128, HDIM);
                tdm_load_2d(buf ? ldsB0 : ldsB1,
                            Wsel + (size_t)nBase * HDIM + k1, HDIM, HD, 32, 64, HDIM);
                __builtin_amdgcn_s_wait_tensorcnt(2);   // pair(it) complete
            } else {
                __builtin_amdgcn_s_wait_tensorcnt(0);
            }
        }
        __syncthreads();   // publish buf to all waves

        // --- A fragments (lane = M row; K halves per ISA 16-bit A layout) ---
        const _Float16* ap0 = &As[buf][wm * 32 + lrow][0];
        const _Float16* ap1 = &As[buf][wm * 32 + 16 + lrow][0];
        v8h a0lo = *(const v8h*)(ap0 + 8 * half);
        v8h a0hi = *(const v8h*)(ap0 + 16 + 8 * half);
        v8h a1lo = *(const v8h*)(ap1 + 8 * half);
        v8h a1hi = *(const v8h*)(ap1 + 16 + 8 * half);
        v16h af0, af1;
        #pragma unroll
        for (int i = 0; i < 8; ++i) {
            af0[i] = a0lo[i]; af0[i + 8] = a0hi[i];
            af1[i] = a1lo[i]; af1[i + 8] = a1hi[i];
        }

        // --- B fragments (lane = N column; Bs is [n][k]) ---
        const _Float16* bp0 = &Bs[buf][wn * 32 + lrow][0];
        const _Float16* bp1 = &Bs[buf][wn * 32 + 16 + lrow][0];
        v8h b0lo = *(const v8h*)(bp0 + 8 * half);
        v8h b0hi = *(const v8h*)(bp0 + 16 + 8 * half);
        v8h b1lo = *(const v8h*)(bp1 + 8 * half);
        v8h b1hi = *(const v8h*)(bp1 + 16 + 8 * half);
        v16h bf0, bf1;
        #pragma unroll
        for (int i = 0; i < 8; ++i) {
            bf0[i] = b0lo[i]; bf0[i + 8] = b0hi[i];
            bf1[i] = b1lo[i]; bf1[i + 8] = b1hi[i];
        }

        acc00 = __builtin_amdgcn_wmma_f32_16x16x32_f16(false, af0, false, bf0, (short)0, acc00, false, false);
        acc01 = __builtin_amdgcn_wmma_f32_16x16x32_f16(false, af0, false, bf1, (short)0, acc01, false, false);
        acc10 = __builtin_amdgcn_wmma_f32_16x16x32_f16(false, af1, false, bf0, (short)0, acc10, false, false);
        acc11 = __builtin_amdgcn_wmma_f32_16x16x32_f16(false, af1, false, bf1, (short)0, acc11, false, false);

        __syncthreads();   // all waves done with buf before wave0 re-fills it
    }

    // --- epilogue: D layout lane=N, VGPR v -> M = v + 8*half ---
    const int ncol0 = nBase + wn * 32 + lrow;
    const int ncol1 = ncol0 + 16;
    const float bias0 = bias ? bias[ncol0] : 0.0f;
    const float bias1 = bias ? bias[ncol1] : 0.0f;
    #pragma unroll
    for (int v = 0; v < 8; ++v) {
        const int m0 = mBase + wm * 32 + v + 8 * half;
        const int m1 = m0 + 16;
        float x00 = acc00[v] + bias0, x01 = acc01[v] + bias1;
        float x10 = acc10[v] + bias0, x11 = acc11[v] + bias1;
        if (mat == 0) {
            x00 = fmaxf(x00, 0.0f); x01 = fmaxf(x01, 0.0f);
            x10 = fmaxf(x10, 0.0f); x11 = fmaxf(x11, 0.0f);
        }
        outp[m0 * HD + ncol0] = x00;
        outp[m0 * HD + ncol1] = x01;
        outp[m1 * HD + ncol0] = x10;
        outp[m1 * HD + ncol1] = x11;
    }
}

// ---------------------------------------------------------------------------
// Kernel 2: start_logits[row] = sum_d h_s[row,d]*w_s2[d] + b_s2
// ---------------------------------------------------------------------------
__global__ __launch_bounds__(256)
void start_logits_k(const float* __restrict__ hs, const float* __restrict__ w_s2,
                    const float* __restrict__ b_s2, float* __restrict__ out)
{
    const int row  = blockIdx.x * 8 + (threadIdx.x >> 5);
    const int lane = threadIdx.x & 31;
    float s = 0.0f;
    #pragma unroll 4
    for (int d = lane; d < HD; d += 32) s += hs[row * HD + d] * w_s2[d];
    #pragma unroll
    for (int off = 16; off > 0; off >>= 1) s += __shfl_xor(s, off, 32);
    if (lane == 0) out[row] = s + b_s2[0];
}

// ---------------------------------------------------------------------------
// Kernel 3: end_logits[b,i,j] = sum_d relu(A[b,i,d]+C[b,j,d])*w_e2[d] + b_e2
// ---------------------------------------------------------------------------
#define DCH 128
#define LDP 132
__global__ __launch_bounds__(256)
void end_logits_k(const float* __restrict__ Ap, const float* __restrict__ Cm,
                  const float* __restrict__ w_e2, const float* __restrict__ b_e2,
                  float* __restrict__ out)
{
    __shared__ float sA[16 * LDP];
    __shared__ float sC[16 * LDP];
    __shared__ float sw[DCH];

    const int tid = threadIdx.x;
    const int jt  = blockIdx.x;   // 0..31
    const int it  = blockIdx.y;   // 0..31
    const int b   = blockIdx.z;   // 0..1

    const int il = tid >> 4, jl = tid & 15;
    const int r  = tid >> 4, seg = tid & 15;

    float4 acc = {0.f, 0.f, 0.f, 0.f};

    for (int c0 = 0; c0 < HD; c0 += DCH) {
        __syncthreads();
        {
            const float* ga = Ap + (size_t)(b * SEQ + it * 16 + r) * HD + c0;
            const float* gc = Cm + (size_t)(b * SEQ + jt * 16 + r) * HD + c0;
            float4 a0 = ((const float4*)ga)[seg];
            float4 a1 = ((const float4*)ga)[seg + 16];
            float4 cA = ((const float4*)gc)[seg];
            float4 cB = ((const float4*)gc)[seg + 16];
            if (c0 + DCH < HD) {
                __builtin_prefetch(ga + DCH, 0, 1);
                __builtin_prefetch(gc + DCH, 0, 1);
            }
            *(float4*)&sA[r * LDP + seg * 4]      = a0;
            *(float4*)&sA[r * LDP + 64 + seg * 4] = a1;
            *(float4*)&sC[r * LDP + seg * 4]      = cA;
            *(float4*)&sC[r * LDP + 64 + seg * 4] = cB;
            if (tid < DCH / 4)
                *(float4*)&sw[tid * 4] = ((const float4*)(w_e2 + c0))[tid];
        }
        __syncthreads();

        const float* ai = &sA[il * LDP];
        const float* cj = &sC[jl * LDP];
        #pragma unroll 8
        for (int kk = 0; kk < DCH; kk += 4) {
            float4 a  = *(const float4*)&ai[kk];
            float4 c  = *(const float4*)&cj[kk];
            float4 wv = *(const float4*)&sw[kk];
            acc.x += fmaxf(a.x + c.x, 0.0f) * wv.x;
            acc.y += fmaxf(a.y + c.y, 0.0f) * wv.y;
            acc.z += fmaxf(a.z + c.z, 0.0f) * wv.z;
            acc.w += fmaxf(a.w + c.w, 0.0f) * wv.w;
        }
    }

    const float res = acc.x + acc.y + acc.z + acc.w + b_e2[0];
    out[(size_t)b * SEQ * SEQ + (size_t)(it * 16 + il) * SEQ + jt * 16 + jl] = res;
}

// ---------------------------------------------------------------------------
extern "C" void kernel_launch(void* const* d_in, const int* in_sizes, int n_in,
                              void* d_out, int out_size, void* d_ws, size_t ws_size,
                              hipStream_t stream)
{
    const float* emb  = (const float*)d_in[0];
    const float* W_s1 = (const float*)d_in[1];
    const float* b_s1 = (const float*)d_in[2];
    const float* w_s2 = (const float*)d_in[3];
    const float* b_s2 = (const float*)d_in[4];
    const float* W_e1 = (const float*)d_in[5];
    const float* b_e1 = (const float*)d_in[6];
    const float* w_e2 = (const float*)d_in[7];
    const float* b_e2 = (const float*)d_in[8];

    float* out       = (float*)d_out;
    float* start_out = out;            // (B,N) = 1024 floats
    float* end_out   = out + MTOT;     // (B,N,N) = 524288 floats

    // workspace layout (11 MB total)
    char* ws = (char*)d_ws;
    _Float16* embh = (_Float16*)(ws);                       // 2 MB  [1024][1024]
    _Float16* wsT  = (_Float16*)(ws + (size_t)(2 << 20));   // 1 MB  [512][1024]
    _Float16* weTs = (_Float16*)(ws + (size_t)(3 << 20));   // 1 MB  [512][1024]
    _Float16* weTe = (_Float16*)(ws + (size_t)(4 << 20));   // 1 MB  [512][1024]
    float*    hs   = (float*)(ws + (size_t)(5 << 20));      // 2 MB
    float*    Ap   = (float*)(ws + (size_t)(7 << 20));      // 2 MB
    float*    Cm   = (float*)(ws + (size_t)(9 << 20));      // 2 MB

    // 0: conversions (emb straight, weights transposed to [n][k])
    cvt_f32_to_f16<<<(MTOT * HDIM / 4 + 255) / 256, 256, 0, stream>>>(emb, embh, MTOT * HDIM / 4);
    {
        dim3 gt(HDIM / 32, HD / 32);   // K=1024 rows, N=512 cols
        cvt_transpose_f16<<<gt, 256, 0, stream>>>(W_s1, wsT, HDIM, HD);
        cvt_transpose_f16<<<gt, 256, 0, stream>>>(W_e1, weTs, HDIM, HD);
        cvt_transpose_f16<<<gt, 256, 0, stream>>>(W_e1 + (size_t)HDIM * HD, weTe, HDIM, HD);
    }

    // 1: fused WMMA projections, TDM-staged
    dim3 g1(MTOT / 128, 24);
    fused_proj_wmma<<<g1, 256, 0, stream>>>(embh, wsT, weTs, weTe, b_s1, b_e1, hs, Ap, Cm);

    // 2: start logits
    start_logits_k<<<MTOT / 8, 256, 0, stream>>>(hs, w_s2, b_s2, start_out);

    // 3: end logits
    dim3 g3(SEQ / 16, SEQ / 16, BATCH);
    end_logits_k<<<g3, 256, 0, stream>>>(Ap, Cm, w_e2, b_e2, end_out);
}
